// SelfAttention_66219805769895
// MI455X (gfx1250) — matrix operand order repaired and checked by reading, not compiled, not run on hardware
//
#include <hip/hip_runtime.h>

// ---------------------------------------------------------------------------
// SelfAttention forward for MI455X (gfx1250, wave32, WMMA).
//
// Roofline: ~103 GFLOP vs ~0.47 GB of mandatory HBM traffic (268 MB of that is
// the att tensor the reference returns). At 23.3 TB/s the att write dominates
// (~20 us); bf16 WMMA compute is far cheaper. Therefore: write att exactly
// once, keep the att tile resident in LDS (bf16) for the att@V GEMM, and run
// every GEMM through v_wmma_f32_16x16x32_bf16 with f32 accumulation.
// ---------------------------------------------------------------------------

constexpr int BATCH = 32;
constexpr int NUM   = 512;
constexpr int FEAT  = 1024;
constexpr int NH    = 8;
constexpr int HLEN  = 128;   // FEAT / NH

typedef __attribute__((ext_vector_type(16))) __bf16 v16b;
typedef __attribute__((ext_vector_type(8)))  __bf16 v8b;
typedef __attribute__((ext_vector_type(8)))  float  v8f;

union Frag { v16b v; v8b h2[2]; };

static __device__ inline __bf16 f2bf(float f) {
  unsigned u = __builtin_bit_cast(unsigned, f);
  u += 0x7FFFu + ((u >> 16) & 1u);          // round-to-nearest-even
  unsigned short s = (unsigned short)(u >> 16);
  return __builtin_bit_cast(__bf16, s);
}

// --------------------------------------------------------------------------
// gate2s[b,f] = (mean_n x2[b,n,f])^2 * (1/sqrt(HLEN))   (softmax scale folded)
// --------------------------------------------------------------------------
__global__ void __launch_bounds__(256)
k_gate(const float* __restrict__ x2, float* __restrict__ g2s) {
  int idx = blockIdx.x * 256 + threadIdx.x;      // b*FEAT + f
  int b = idx >> 10;
  int f = idx & (FEAT - 1);
  const float* p = x2 + (size_t)b * NUM * FEAT + f;
  float s = 0.0f;
#pragma unroll 4
  for (int n = 0; n < NUM; ++n) s += p[(size_t)n * FEAT];
  s *= (1.0f / (float)NUM);
  g2s[idx] = s * s * 0.08838834764831845f;       // 1/sqrt(128)
}

// --------------------------------------------------------------------------
// elementwise fp32 -> bf16
// --------------------------------------------------------------------------
__global__ void __launch_bounds__(256)
k_cvt(const float* __restrict__ src, __bf16* __restrict__ dst, int n) {
  int i = blockIdx.x * 256 + threadIdx.x;
  if (i < n) dst[i] = f2bf(src[i]);
}

// --------------------------------------------------------------------------
// batched tiled transpose: dst[c,r] = bf16(src[r,c]); dims multiple of 32
// --------------------------------------------------------------------------
__global__ void __launch_bounds__(256)
k_transpose(const float* __restrict__ src, __bf16* __restrict__ dst, int R, int C) {
  __shared__ float tile[32][33];
  int batch = blockIdx.z;
  int c0 = blockIdx.x * 32, r0 = blockIdx.y * 32;
  const float* s = src + (size_t)batch * R * C;
  __bf16* d = dst + (size_t)batch * R * C;
  int tx = threadIdx.x, ty = threadIdx.y;        // 32 x 8
#pragma unroll
  for (int j = 0; j < 32; j += 8)
    tile[ty + j][tx] = s[(size_t)(r0 + ty + j) * C + (c0 + tx)];
  __syncthreads();
#pragma unroll
  for (int j = 0; j < 32; j += 8)
    d[(size_t)(c0 + ty + j) * R + (r0 + tx)] = f2bf(tile[tx][ty + j]);
}

// --------------------------------------------------------------------------
// Q projection: Qs[b,n,h*128+d] = bf16((x1 @ Wq[h] + bq) * g2s) ; one 16x16
// output tile per wave, K = 1024 via 32 WMMA k-steps.
// --------------------------------------------------------------------------
__global__ void __launch_bounds__(128)
k_qproj(const __bf16* __restrict__ x1bf, const __bf16* __restrict__ WqT,
        const float* __restrict__ bq, const float* __restrict__ g2s,
        __bf16* __restrict__ Qs) {
  __shared__ __bf16 stage[4][16 * 16];
  int lane = threadIdx.x & 31, wave = threadIdx.x >> 5;
  int tile = blockIdx.x * 4 + wave;              // b(32) x mt(32) x nt(64)
  int nt = tile & 63; tile >>= 6;
  int mt = tile & 31; int b = tile >> 5;
  int hl = lane >> 4, l = lane & 15;
  int h = nt >> 3;                               // 8 column-tiles per head
  int dcol = ((nt & 7) << 4) + l;                // d index of this lane's column

  const __bf16* arow = x1bf + (size_t)(b * NUM + mt * 16 + l) * FEAT;
  const __bf16* brow = WqT + ((size_t)h * HLEN + dcol) * FEAT;
  v8f acc = {};
  for (int kc = 0; kc < FEAT / 32; ++kc) {
    int k0 = kc * 32;
    Frag A, Bf;
    A.h2[0]  = *(const v8b*)(arow + k0 + hl * 8);
    A.h2[1]  = *(const v8b*)(arow + k0 + 16 + hl * 8);
    Bf.h2[0] = *(const v8b*)(brow + k0 + hl * 8);
    Bf.h2[1] = *(const v8b*)(brow + k0 + 16 + hl * 8);
    acc = __builtin_amdgcn_wmma_f32_16x16x32_bf16(false, A.v, false, Bf.v,
                                                  (short)0, acc, false, false);
  }
  float bqv = bq[h * HLEN + dcol];
  float gv  = g2s[b * FEAT + h * HLEN + dcol];   // gate^2 * softmax scale
  __bf16* st = stage[wave];
#pragma unroll
  for (int j = 0; j < 8; ++j)
    st[(j + hl * 8) * 16 + l] = f2bf((acc[j] + bqv) * gv);
  __syncthreads();
  int row = lane >> 1, part = lane & 1;
  __bf16* drow = Qs + (size_t)(b * NUM + mt * 16 + row) * FEAT +
                 h * HLEN + ((nt & 7) << 4) + part * 8;
  *(v8b*)drow = *(const v8b*)(st + row * 16 + part * 8);
}

// --------------------------------------------------------------------------
// Fused attention: per wave, one (b, h, 16-row tile).
//   aff = Qs @ K^T (WMMA, gate/scale pre-folded) -> mask -> softmax ->
//   write att fp32 to d_out + keep bf16 copy in LDS -> V_ = att @ V (WMMA).
// LDS per wave: 16x512 f32 (aff) + 16x512 bf16 (att) + 16x16 bf16 staging.
// --------------------------------------------------------------------------
constexpr int WAVE_LDS = 16 * NUM * 4 + 16 * NUM * 2 + 16 * 16 * 2;  // 49664 B

__global__ void __launch_bounds__(64)
k_attn(const __bf16* __restrict__ Qs, const __bf16* __restrict__ x1bf,
       const __bf16* __restrict__ x1T, const unsigned char* __restrict__ mask,
       float* __restrict__ att_out, __bf16* __restrict__ Vbuf) {
  extern __shared__ char smem[];
  int lane = threadIdx.x & 31, wave = threadIdx.x >> 5;
  char* base = smem + (size_t)wave * WAVE_LDS;
  float*  aff  = (float*)base;                         // 16 x 512 f32
  __bf16* attb = (__bf16*)(base + 16 * NUM * 4);       // 16 x 512 bf16
  __bf16* stg  = (__bf16*)(base + 16 * NUM * 4 + 16 * NUM * 2);  // 16x16

  int tile = blockIdx.x * 2 + wave;                    // b(32) x h(8) x mt(32)
  int mt = tile & 31; tile >>= 5;
  int h = tile & 7; int b = tile >> 3;
  int hl = lane >> 4, l = lane & 15;

  // A fragments: 16 rows of Qs, full K=128 (4 k-chunks), kept in VGPRs.
  Frag afrag[4];
  const __bf16* qrow = Qs + (size_t)(b * NUM + mt * 16 + l) * FEAT + h * HLEN;
#pragma unroll
  for (int kc = 0; kc < 4; ++kc) {
    afrag[kc].h2[0] = *(const v8b*)(qrow + kc * 32 + hl * 8);
    afrag[kc].h2[1] = *(const v8b*)(qrow + kc * 32 + 16 + hl * 8);
  }

  const unsigned char* mrow = mask + (size_t)b * NUM * NUM;

  // ---- QK^T over 32 column tiles; masked logits into LDS ----
  for (int nt = 0; nt < 32; ++nt) {
    const __bf16* krow = x1bf + (size_t)(b * NUM + nt * 16 + l) * FEAT + h * HLEN;
    if (nt + 1 < 32)
      __builtin_prefetch(x1bf + (size_t)(b * NUM + (nt + 1) * 16 + l) * FEAT + h * HLEN, 0, 3);
    v8f acc = {};
#pragma unroll
    for (int kc = 0; kc < 4; ++kc) {
      Frag Bf;
      Bf.h2[0] = *(const v8b*)(krow + kc * 32 + hl * 8);
      Bf.h2[1] = *(const v8b*)(krow + kc * 32 + 16 + hl * 8);
      acc = __builtin_amdgcn_wmma_f32_16x16x32_bf16(false, afrag[kc].v, false, Bf.v,
                                                    (short)0, acc, false, false);
    }
#pragma unroll
    for (int j = 0; j < 8; ++j) {
      int M = j + hl * 8;
      int col = nt * 16 + l;
      unsigned char mk = mrow[(size_t)(mt * 16 + M) * NUM + col];
      aff[M * NUM + col] = mk ? -1e9f : acc[j];
    }
  }
  __syncthreads();

  // ---- row softmax (16 rows x 512), att -> global fp32 + LDS bf16 ----
  for (int r = 0; r < 16; ++r) {
    float* row = aff + r * NUM;
    float mx = -3.0e38f;
    for (int i = lane; i < NUM; i += 32) mx = fmaxf(mx, row[i]);
#pragma unroll
    for (int off = 16; off > 0; off >>= 1) mx = fmaxf(mx, __shfl_xor(mx, off, 32));
    float sum = 0.0f;
    for (int i = lane; i < NUM; i += 32) {
      float e = __expf(row[i] - mx);
      row[i] = e;
      sum += e;
    }
#pragma unroll
    for (int off = 16; off > 0; off >>= 1) sum += __shfl_xor(sum, off, 32);
    float inv = 1.0f / sum;
    float* gout = att_out + (((size_t)(b * NH + h) * NUM + mt * 16 + r) * NUM);
    for (int i = lane; i < NUM; i += 32) {
      float a = row[i] * inv;
      gout[i] = a;
      attb[r * NUM + i] = f2bf(a);
    }
  }
  __syncthreads();

  // ---- V_ = att @ V ; att (A) from LDS bf16, V^T (B) from x1T ----
  for (int dt = 0; dt < 8; ++dt) {
    const __bf16* vrow = x1T + ((size_t)b * FEAT + h * HLEN + dt * 16 + l) * NUM;
    v8f acc = {};
    for (int kc = 0; kc < 16; ++kc) {
      Frag A2, B2;
      A2.h2[0] = *(const v8b*)(attb + l * NUM + kc * 32 + hl * 8);
      A2.h2[1] = *(const v8b*)(attb + l * NUM + kc * 32 + 16 + hl * 8);
      B2.h2[0] = *(const v8b*)(vrow + kc * 32 + hl * 8);
      B2.h2[1] = *(const v8b*)(vrow + kc * 32 + 16 + hl * 8);
      acc = __builtin_amdgcn_wmma_f32_16x16x32_bf16(false, A2.v, false, B2.v,
                                                    (short)0, acc, false, false);
    }
#pragma unroll
    for (int j = 0; j < 8; ++j) stg[(j + hl * 8) * 16 + l] = f2bf(acc[j]);
    __syncthreads();
    int row = lane >> 1, part = lane & 1;
    __bf16* drow = Vbuf + (size_t)(b * NUM + mt * 16 + row) * FEAT +
                   h * HLEN + dt * 16 + part * 8;
    *(v8b*)drow = *(const v8b*)(stg + row * 16 + part * 8);
    __syncthreads();
  }
}

// --------------------------------------------------------------------------
// xout = relu(V_ @ Wl + bl) + x1 ; one 16x16 tile per wave, K = 1024.
// --------------------------------------------------------------------------
__global__ void __launch_bounds__(128)
k_out(const __bf16* __restrict__ Vbuf, const __bf16* __restrict__ WlT,
      const float* __restrict__ bl, const float* __restrict__ x1,
      float* __restrict__ xout) {
  __shared__ float stage[4][16 * 16];
  int lane = threadIdx.x & 31, wave = threadIdx.x >> 5;
  int tile = blockIdx.x * 4 + wave;              // b(32) x mt(32) x nt(64)
  int nt = tile & 63; tile >>= 6;
  int mt = tile & 31; int b = tile >> 5;
  int hl = lane >> 4, l = lane & 15;

  const __bf16* arow = Vbuf + (size_t)(b * NUM + mt * 16 + l) * FEAT;
  const __bf16* brow = WlT + (size_t)(nt * 16 + l) * FEAT;
  v8f acc = {};
  for (int kc = 0; kc < FEAT / 32; ++kc) {
    int k0 = kc * 32;
    Frag A, Bf;
    A.h2[0]  = *(const v8b*)(arow + k0 + hl * 8);
    A.h2[1]  = *(const v8b*)(arow + k0 + 16 + hl * 8);
    Bf.h2[0] = *(const v8b*)(brow + k0 + hl * 8);
    Bf.h2[1] = *(const v8b*)(brow + k0 + 16 + hl * 8);
    acc = __builtin_amdgcn_wmma_f32_16x16x32_bf16(false, A.v, false, Bf.v,
                                                  (short)0, acc, false, false);
  }
  float blv = bl[nt * 16 + l];
  float* st = stage[wave];
#pragma unroll
  for (int j = 0; j < 8; ++j)
    st[(j + hl * 8) * 16 + l] = fmaxf(acc[j] + blv, 0.0f);
  __syncthreads();
  int row = lane >> 1, part = lane & 1;
  size_t off = (size_t)(b * NUM + mt * 16 + row) * FEAT + nt * 16 + part * 8;
  const float* xr = x1 + off;
  float* orow = xout + off;
#pragma unroll
  for (int q = 0; q < 8; ++q) orow[q] = st[row * 16 + part * 8 + q] + xr[q];
}

// --------------------------------------------------------------------------
extern "C" void kernel_launch(void* const* d_in, const int* in_sizes, int n_in,
                              void* d_out, int out_size, void* d_ws, size_t ws_size,
                              hipStream_t stream) {
  (void)in_sizes; (void)n_in; (void)out_size; (void)ws_size;
  const float* x1 = (const float*)d_in[0];
  const float* x2 = (const float*)d_in[1];
  // d_in[2] (adj) and d_in[3] (bias) are unused by the reference forward.
  const unsigned char* mask = (const unsigned char*)d_in[4];  // jax bool = 1 byte
  const float* Wq = (const float*)d_in[5];
  const float* bq = (const float*)d_in[6];
  const float* Wl = (const float*)d_in[7];
  const float* bl = (const float*)d_in[8];

  float* xout = (float*)d_out;
  float* att  = xout + (size_t)BATCH * NUM * FEAT;

  char* ws = (char*)d_ws;
  float*  g2s  = (float*)ws;  ws += (size_t)BATCH * FEAT * 4;
  __bf16* x1bf = (__bf16*)ws; ws += (size_t)BATCH * NUM * FEAT * 2;
  __bf16* x1T  = (__bf16*)ws; ws += (size_t)BATCH * NUM * FEAT * 2;
  __bf16* Qs   = (__bf16*)ws; ws += (size_t)BATCH * NUM * FEAT * 2;
  __bf16* Vbuf = (__bf16*)ws; ws += (size_t)BATCH * NUM * FEAT * 2;
  __bf16* WqT  = (__bf16*)ws; ws += (size_t)NH * HLEN * FEAT * 2;
  __bf16* WlT  = (__bf16*)ws; ws += (size_t)FEAT * FEAT * 2;

  k_gate<<<BATCH * FEAT / 256, 256, 0, stream>>>(x2, g2s);
  k_cvt<<<(BATCH * NUM * FEAT) / 256, 256, 0, stream>>>(x1, x1bf, BATCH * NUM * FEAT);
  k_transpose<<<dim3(FEAT / 32, NUM / 32, BATCH), dim3(32, 8), 0, stream>>>(x1, x1T, NUM, FEAT);
  k_transpose<<<dim3(HLEN / 32, FEAT / 32, NH), dim3(32, 8), 0, stream>>>(Wq, WqT, FEAT, HLEN);
  k_transpose<<<dim3(FEAT / 32, FEAT / 32, 1), dim3(32, 8), 0, stream>>>(Wl, WlT, FEAT, FEAT);

  k_qproj<<<(BATCH * 32 * 64) / 4, 128, 0, stream>>>(x1bf, WqT, bq, g2s, Qs);

  size_t attn_smem = 2 * (size_t)WAVE_LDS;   // 2 waves/block, 99,328 B
  k_attn<<<(BATCH * NH * 32) / 2, 64, attn_smem, stream>>>(Qs, x1bf, x1T, mask, att, Vbuf);

  k_out<<<(BATCH * 32 * 64) / 4, 128, 0, stream>>>(Vbuf, WlT, bl, x1, xout);
}
